// QAttention_32744830665502
// MI455X (gfx1250) — matrix operand order repaired and checked by reading, not compiled, not run on hardware
//
#include <hip/hip_runtime.h>
#include <hip/hip_bf16.h>

// ---------------------------------------------------------------------------
// Quaternion attention for MI455X (gfx1250): all GEMM-like work on
// v_wmma_f32_16x16x32_bf16; flash-attention with LDS-staged K/V tiles,
// WMMA-based row-sum, scalar (SGPR) causal guard, CDNA5 split waits.
// ---------------------------------------------------------------------------

typedef __attribute__((ext_vector_type(16))) __bf16 v16bf;
typedef __attribute__((ext_vector_type(8)))  __bf16 v8bf;
typedef __attribute__((ext_vector_type(8)))  float  v8f;

#define NB_  2
#define NS_  2048
#define NDIM_ 1024
#define NH_  16
#define NDH_ 64
#define NG_  16
#define NTOK_ (NB_*NS_)   // 4096

static __device__ __forceinline__ v16bf join8(v8bf lo, v8bf hi){
  v16bf r;
#pragma unroll
  for(int i=0;i<8;i++){ r[i]=lo[i]; r[i+8]=hi[i]; }
  return r;
}

static __device__ __forceinline__ v8f zero8(){
  v8f z;
#pragma unroll
  for(int i=0;i<8;i++) z[i]=0.0f;
  return z;
}

// A fragment 16x32 (MxK) bf16: lane holds row M=lane%16; K-base=(lane/16)*8,
// elements 0..7 at k0+kbase, elements 8..15 at k0+16+kbase. (ISA 7.12.2)
static __device__ __forceinline__ v16bf load_a_frag(const __bf16* __restrict__ base,
                                                    int row, int ld, int k0, int lane){
  const __bf16* p = base + (size_t)row*ld + k0 + ((lane>>4)<<3);
  v8bf lo = *(const v8bf*)(p);
  v8bf hi = *(const v8bf*)(p + 16);
  return join8(lo,hi);
}

// B fragment 32x16 (KxN) bf16: lane holds col N=lane%16; K-base=(lane/16)*8,
// elements 0..7 at k0+kbase, elements 8..15 at k0+16+kbase. Source row-major [N][K].
static __device__ __forceinline__ v16bf load_b_frag(const __bf16* __restrict__ base,
                                                    int n, int ld, int k0, int lane){
  const __bf16* p = base + (size_t)n*ld + k0 + ((lane>>4)<<3);
  v8bf lo = *(const v8bf*)(p);
  v8bf hi = *(const v8bf*)(p + 16);
  return join8(lo,hi);
}

// Same B-fragment shape but from an LDS tile row.
static __device__ __forceinline__ v16bf lds_b_frag(const __bf16* p0, int lane){
  const __bf16* p = p0 + ((lane>>4)<<3);
  v8bf lo = *(const v8bf*)(p);
  v8bf hi = *(const v8bf*)(p + 16);
  return join8(lo,hi);
}

#define WMMA_BF16(a,b,c) \
  __builtin_amdgcn_wmma_f32_16x16x32_bf16(false,(a),false,(b),(short)0,(c),false,false)

// ---------------------------------------------------------------------------
// fp32 -> bf16 convert
// ---------------------------------------------------------------------------
__global__ __launch_bounds__(256) void cvt_f32_bf16(const float* __restrict__ in,
                                                    __bf16* __restrict__ out, int n){
  int i = blockIdx.x*256 + threadIdx.x;
  if (i < n) out[i] = (__bf16)in[i];
}

// ---------------------------------------------------------------------------
// Expand quaternion weights (wr,wi,wj,wk each [256][256]) into a dense
// bf16 [1024][1024] matrix Wbig[out_ch][in_ch] implementing qlinear.
// ---------------------------------------------------------------------------
__global__ __launch_bounds__(256) void expand_qweights(const float* __restrict__ wr,
                                                       const float* __restrict__ wi,
                                                       const float* __restrict__ wj,
                                                       const float* __restrict__ wk,
                                                       __bf16* __restrict__ Wbig){
  int idx = blockIdx.x*256 + threadIdx.x;           // 1024*1024 entries
  if (idx >= NDIM_*NDIM_) return;
  int nb = idx >> 10, kb = idx & 1023;
  int m = nb >> 2, c = nb & 3;
  int n = kb >> 2, d = kb & 3;
  int off = m*256 + n;
  const float* srcs[4][4] = {
    {wr, wi, wj, wk},
    {wi, wr, wj, wk},
    {wj, wi, wr, wk},
    {wk, wi, wj, wr}};
  const float signs[4][4] = {
    {1.f,-1.f,-1.f,-1.f},
    {1.f, 1.f,-1.f, 1.f},
    {1.f, 1.f, 1.f,-1.f},
    {1.f,-1.f, 1.f, 1.f}};
  Wbig[idx] = (__bf16)(signs[c][d] * srcs[c][d][off]);
}

// ---------------------------------------------------------------------------
// bf16 WMMA GEMM: Y[M][N] = X[M][K] * W[N][K]^T + bias
// Block tile 128x128, 8 waves (4 along M x 2 along N), each wave 32x64
// -> 2x4 accumulators, 8 v_wmma per K-step of 32. Prefetch next K panel.
// ---------------------------------------------------------------------------
template<bool OUT_F32>
__global__ __launch_bounds__(256)
void qgemm(const __bf16* __restrict__ X, const __bf16* __restrict__ W,
           const float* __restrict__ bias, void* __restrict__ Yout,
           int M, int N, int K){
  const int lane = threadIdx.x & 31;
  const int wave = __builtin_amdgcn_readfirstlane(threadIdx.x >> 5);
  const int wm = wave & 3;
  const int wn = wave >> 2;
  const int rowBase = blockIdx.y*128 + wm*32;
  const int colBase = blockIdx.x*128 + wn*64;
  const int hi = lane >> 4, ln = lane & 15;

  v8f acc[2][4];
#pragma unroll
  for(int i=0;i<2;i++)
#pragma unroll
    for(int j=0;j<4;j++) acc[i][j] = zero8();

  for(int k0=0;k0<K;k0+=32){
    if (k0 + 32 < K){
      __builtin_prefetch((const void*)(X + (size_t)(rowBase + ln)*K + k0 + 32), 0, 1);
      __builtin_prefetch((const void*)(W + (size_t)(colBase + ln)*K + k0 + 32), 0, 1);
    }
    v16bf a[2], b[4];
#pragma unroll
    for(int mf=0;mf<2;mf++)
      a[mf] = load_a_frag(X, rowBase + mf*16 + ln, K, k0, lane);
#pragma unroll
    for(int nf=0;nf<4;nf++)
      b[nf] = load_b_frag(W, colBase + nf*16 + ln, K, k0, lane);
#pragma unroll
    for(int mf=0;mf<2;mf++)
#pragma unroll
      for(int nf=0;nf<4;nf++)
        acc[mf][nf] = WMMA_BF16(a[mf], b[nf], acc[mf][nf]);
  }

  // C/D layout: VGPR v, lane l -> M = v + 8*(l/16), N = l%16
#pragma unroll
  for(int mf=0;mf<2;mf++){
#pragma unroll
    for(int nf=0;nf<4;nf++){
#pragma unroll
      for(int v=0;v<8;v++){
        int r = rowBase + mf*16 + v + 8*hi;
        int c = colBase + nf*16 + ln;
        float val = acc[mf][nf][v] + bias[c];
        if (OUT_F32) ((float*)Yout)[(size_t)r*N + c] = val;
        else         ((__bf16*)Yout)[(size_t)r*N + c] = (__bf16)val;
      }
    }
  }
}

// ---------------------------------------------------------------------------
// Quaternion RoPE, in-place on [B*S][1024] bf16 (layout n = h*64 + g*4 + comp)
// ---------------------------------------------------------------------------
__global__ __launch_bounds__(256) void qrope_kernel(__bf16* __restrict__ Y, int total){
  int idx = blockIdx.x*256 + threadIdx.x;            // B*S*H*G
  if (idx >= total) return;
  int g = idx & (NG_-1);
  int h = (idx >> 4) & (NH_-1);
  int t = idx >> 8;
  int s = t & (NS_-1);
  float inv = __expf(-(float)g * (9.210340371976184f / (float)NG_)); // base^(-g/G)
  float ang = (float)s * inv;
  float sn, cs; __sincosf(ang, &sn, &cs);
  __bf16* p = Y + (size_t)t*NDIM_ + h*NDH_ + g*4;
  float xr=(float)p[0], xi=(float)p[1], xj=(float)p[2], xk=(float)p[3];
  float pr,pi,pj,pk;
  int ax = g % 3;
  if (ax==0)      { pr=cs*xr-sn*xi; pi=cs*xi+sn*xr; pj=cs*xj-sn*xk; pk=cs*xk+sn*xj; }
  else if (ax==1) { pr=cs*xr-sn*xj; pi=cs*xi+sn*xk; pj=cs*xj+sn*xr; pk=cs*xk-sn*xi; }
  else            { pr=cs*xr-sn*xk; pi=cs*xi-sn*xj; pj=cs*xj+sn*xi; pk=cs*xk+sn*xr; }
  p[0]=(__bf16)pr; p[1]=(__bf16)pi; p[2]=(__bf16)pj; p[3]=(__bf16)pk;
}

// ---------------------------------------------------------------------------
// V transpose: [B*S][H*64] -> Vt[B][H][64][S]
// ---------------------------------------------------------------------------
__global__ __launch_bounds__(256) void transpose_v(const __bf16* __restrict__ V,
                                                   __bf16* __restrict__ Vt){
  int idx = blockIdx.x*256 + threadIdx.x;            // B*H*64*S elements
  if (idx >= NB_*NH_*NDH_*NS_) return;
  int s = idx & (NS_-1);
  int d = (idx >> 11) & (NDH_-1);
  int h = (idx >> 17) & (NH_-1);
  int b = idx >> 21;
  Vt[idx] = V[((size_t)(b*NS_ + s))*NDIM_ + h*NDH_ + d];
}

// ---------------------------------------------------------------------------
// Flash attention, causal. Block = 4 waves x 16 queries (64-query block),
// key tiles of 32 staged cooperatively in LDS (coalesced once, reused 4x).
// Per tile / wave: 4 score WMMAs + 1 row-sum WMMA (P*ones) + 4 AV WMMAs.
// Wave id / causal bound forced into SGPRs (readfirstlane) so the per-tile
// skip is a scalar branch: EXEC provably stays all-ones around WMMAs.
// ---------------------------------------------------------------------------
__global__ __launch_bounds__(128)
void qflash(const __bf16* __restrict__ Q, const __bf16* __restrict__ Km,
            const __bf16* __restrict__ Vt, __bf16* __restrict__ O){
  __shared__ __bf16 kt_s[32][64];     // [key][d]   4 KB
  __shared__ __bf16 vt_s[64][32];     // [d][key]   4 KB
  __shared__ __bf16 pbuf[4][16][32];  // per-wave P 4 KB
  const int tid  = threadIdx.x;
  const int lane = tid & 31;
  const int wave = __builtin_amdgcn_readfirstlane(tid >> 5);  // SGPR
  const int hi = lane >> 4, ln = lane & 15;
  const int bh = blockIdx.y;                // b*H + h
  const int b  = bh >> 4, h = bh & 15;
  const int s0 = blockIdx.x*64 + wave*16;   // wave-uniform (SGPR)

  const __bf16* Qb = Q  + (size_t)b*NS_*NDIM_ + h*NDH_;   // row stride NDIM_
  const __bf16* Kb = Km + (size_t)b*NS_*NDIM_ + h*NDH_;
  const __bf16* Vb = Vt + (size_t)bh*NDH_*NS_;            // [64][S]

  // Q tile 16x64 as two 16x32 A-fragments (loaded once)
  v16bf qa[2];
#pragma unroll
  for(int c=0;c<2;c++)
    qa[c] = load_a_frag(Qb, s0 + ln, NDIM_, c*32, lane);

  // ones B-fragment for WMMA row sums
  v16bf ones;
#pragma unroll
  for(int i=0;i<16;i++) ones[i] = (__bf16)1.0f;

  v8f o[4];
#pragma unroll
  for(int df=0;df<4;df++) o[df] = zero8();
  float m_r[8], l_r[8];
#pragma unroll
  for(int v=0;v<8;v++){ m_r[v] = -1e30f; l_r[v] = 0.0f; }

  const int my_ntiles  = (s0 + 16 + 31) >> 5;            // SGPR, keys 0..s0+15
  const int blk_ntiles = (blockIdx.x*64 + 64 + 31) >> 5; // uniform for block
  for(int tIdx=0; tIdx<blk_ntiles; tIdx++){
    const int kt0 = tIdx*32;

    // ---- cooperative LDS staging (coalesced) ----
    __syncthreads();
    {
      int row = tid >> 2, ch = tid & 3;        // 32 keys x 128B rows
      const v8bf* src = (const v8bf*)(Kb + (size_t)(kt0+row)*NDIM_ + ch*16);
      *(v8bf*)(&kt_s[row][ch*16])     = src[0];
      *(v8bf*)(&kt_s[row][ch*16 + 8]) = src[1];
    }
    {
      int row = tid >> 1, hh = tid & 1;        // 64 d x 64B rows
      const v8bf* src = (const v8bf*)(Vb + (size_t)row*NS_ + kt0 + hh*16);
      *(v8bf*)(&vt_s[row][hh*16])     = src[0];
      *(v8bf*)(&vt_s[row][hh*16 + 8]) = src[1];
    }
    __syncthreads();

    if (tIdx < my_ntiles){          // scalar branch (both sides SGPR)
      // ---- scores: two 16x16 f32 tiles over 32 keys ----
      v8f sfr[2]; sfr[0] = zero8(); sfr[1] = zero8();
#pragma unroll
      for(int nh=0;nh<2;nh++){
#pragma unroll
        for(int c=0;c<2;c++){
          v16bf kf = lds_b_frag(&kt_s[nh*16 + ln][c*32], lane);
          sfr[nh] = WMMA_BF16(qa[c], kf, sfr[nh]);
        }
      }

      // ---- scale + causal mask + online max (shuffle reduce) ----
      float alpha[8];
#pragma unroll
      for(int v=0;v<8;v++){
        int row = s0 + v + 8*hi;
        float x0 = sfr[0][v]*0.125f; if (kt0 + ln      > row) x0 = -1e30f;
        float x1 = sfr[1][v]*0.125f; if (kt0 + 16 + ln > row) x1 = -1e30f;
        sfr[0][v]=x0; sfr[1][v]=x1;
        float mx = fmaxf(x0,x1);
#pragma unroll
        for(int off=1; off<16; off<<=1) mx = fmaxf(mx, __shfl_xor(mx, off, 32));
        float mnew = fmaxf(m_r[v], mx);
        alpha[v] = __expf(m_r[v] - mnew);
        m_r[v] = mnew;
      }

      // ---- P = exp(S - m), rescale O, stage P to LDS ----
#pragma unroll
      for(int v=0;v<8;v++){
        float p0 = __expf(sfr[0][v] - m_r[v]);
        float p1 = __expf(sfr[1][v] - m_r[v]);
#pragma unroll
        for(int df=0; df<4; df++) o[df][v] = o[df][v]*alpha[v];
        int row = v + 8*hi;
        pbuf[wave][row][ln]      = (__bf16)p0;
        pbuf[wave][row][16 + ln] = (__bf16)p1;
      }
      asm volatile("s_wait_dscnt 0" ::: "memory");   // intra-wave LDS flush

      // ---- re-layout P (16x32) as A-fragment from LDS ----
      v16bf pa;
      {
        const __bf16* pr = &pbuf[wave][ln][0];
        int kb = hi*8;
        v8bf lo = *(const v8bf*)(pr + kb);
        v8bf hh = *(const v8bf*)(pr + 16 + kb);
        pa = join8(lo, hh);
      }

      // ---- row sums via WMMA (P * ones): no cross-lane shuffles ----
      v8f rs = WMMA_BF16(pa, ones, zero8());
#pragma unroll
      for(int v=0;v<8;v++) l_r[v] = l_r[v]*alpha[v] + rs[v];

      // ---- O += P * V ----
#pragma unroll
      for(int df=0; df<4; df++){
        v16bf vf = lds_b_frag(&vt_s[df*16 + ln][0], lane);
        o[df] = WMMA_BF16(pa, vf, o[df]);
      }
    }
  }

  // ---- normalize and store [B*S][1024] bf16 ----
#pragma unroll
  for(int df=0; df<4; df++){
#pragma unroll
    for(int v=0; v<8; v++){
      int row = s0 + v + 8*hi;
      int col = h*NDH_ + df*16 + ln;
      O[((size_t)(b*NS_ + row))*NDIM_ + col] = (__bf16)(o[df][v] / l_r[v]);
    }
  }
}

// ---------------------------------------------------------------------------
// Host-side orchestration
// ---------------------------------------------------------------------------
extern "C" void kernel_launch(void* const* d_in, const int* in_sizes, int n_in,
                              void* d_out, int out_size, void* d_ws, size_t ws_size,
                              hipStream_t stream) {
  (void)in_sizes; (void)n_in; (void)out_size; (void)ws_size;

  const float* x   = (const float*)d_in[0];
  // d_in[1] = mask (causal, handled analytically)
  const float* qW[4] = {(const float*)d_in[2],(const float*)d_in[3],(const float*)d_in[4],(const float*)d_in[5]};
  const float* qb  = (const float*)d_in[6];
  const float* kW[4] = {(const float*)d_in[7],(const float*)d_in[8],(const float*)d_in[9],(const float*)d_in[10]};
  const float* kb  = (const float*)d_in[11];
  const float* vW[4] = {(const float*)d_in[12],(const float*)d_in[13],(const float*)d_in[14],(const float*)d_in[15]};
  const float* vb  = (const float*)d_in[16];
  const float* oW[4] = {(const float*)d_in[17],(const float*)d_in[18],(const float*)d_in[19],(const float*)d_in[20]};
  const float* ob  = (const float*)d_in[21];

  char* ws = (char*)d_ws;
  const size_t MB = 1u<<20;
  __bf16* Xbf  = (__bf16*)(ws +  0*MB);   // 8 MB  : [4096][1024]
  __bf16* Wq   = (__bf16*)(ws +  8*MB);   // 2 MB  : [1024][1024]
  __bf16* Wk   = (__bf16*)(ws + 10*MB);
  __bf16* Wv   = (__bf16*)(ws + 12*MB);
  __bf16* Wo   = (__bf16*)(ws + 14*MB);
  __bf16* Qy   = (__bf16*)(ws + 16*MB);   // 8 MB  : [4096][1024]
  __bf16* Ky   = (__bf16*)(ws + 24*MB);
  __bf16* Vy   = (__bf16*)(ws + 32*MB);
  __bf16* Vt   = (__bf16*)(ws + 40*MB);   // 8 MB  : [B][H][64][S]
  __bf16* Attn = (__bf16*)(ws + 48*MB);   // 8 MB  : [4096][1024]

  const int NX = NTOK_*NDIM_;             // 4,194,304
  cvt_f32_bf16<<<(NX+255)/256, 256, 0, stream>>>(x, Xbf, NX);

  expand_qweights<<<(NDIM_*NDIM_+255)/256, 256, 0, stream>>>(qW[0],qW[1],qW[2],qW[3], Wq);
  expand_qweights<<<(NDIM_*NDIM_+255)/256, 256, 0, stream>>>(kW[0],kW[1],kW[2],kW[3], Wk);
  expand_qweights<<<(NDIM_*NDIM_+255)/256, 256, 0, stream>>>(vW[0],vW[1],vW[2],vW[3], Wv);
  expand_qweights<<<(NDIM_*NDIM_+255)/256, 256, 0, stream>>>(oW[0],oW[1],oW[2],oW[3], Wo);

  dim3 gGrid(NDIM_/128, NTOK_/128);       // (8, 32)
  qgemm<false><<<gGrid, 256, 0, stream>>>(Xbf, Wq, qb, Qy, NTOK_, NDIM_, NDIM_);
  qgemm<false><<<gGrid, 256, 0, stream>>>(Xbf, Wk, kb, Ky, NTOK_, NDIM_, NDIM_);
  qgemm<false><<<gGrid, 256, 0, stream>>>(Xbf, Wv, vb, Vy, NTOK_, NDIM_, NDIM_);

  const int NR = NTOK_*NH_*NG_;           // 1,048,576
  qrope_kernel<<<(NR+255)/256, 256, 0, stream>>>(Qy, NR);
  qrope_kernel<<<(NR+255)/256, 256, 0, stream>>>(Ky, NR);

  transpose_v<<<(NX+255)/256, 256, 0, stream>>>(Vy, Vt);

  dim3 fGrid(NS_/64, NB_*NH_);            // (32, 32) x 128 threads
  qflash<<<fGrid, 128, 0, stream>>>(Qy, Ky, Vt, Attn);

  qgemm<true><<<gGrid, 256, 0, stream>>>(Attn, Wo, ob, d_out, NTOK_, NDIM_, NDIM_);
}
